// RecurrentGCN_56238301774487
// MI455X (gfx1250) — matrix operand order repaired and checked by reading, not compiled, not run on hardware
//
#include <hip/hip_runtime.h>
#include <hip/hip_bf16.h>

typedef __attribute__((ext_vector_type(16))) __bf16 v16bf;
typedef __attribute__((ext_vector_type(8)))  float  v8f;

#define F_IN 128
#define HID1 32
#define HID2 16
#define NCLS 10
#define KCH  5

// ---------------- small utility kernels ----------------

__global__ void k_zero(float* p, int n) {
    int i = blockIdx.x * blockDim.x + threadIdx.x;
    if (i < n) p[i] = 0.f;
}

// t = negate ? -s : 0   (prepares T_next = -T_{k-2}, or zeros for k=1)
__global__ void k_prep(float* __restrict__ t, const float* __restrict__ s,
                       int negate, int n) {
    int i = blockIdx.x * blockDim.x + threadIdx.x;
    if (i < n) t[i] = negate ? -s[i] : 0.f;
}

__global__ void k_deg(const int* __restrict__ row, const float* __restrict__ w,
                      float* deg, int E) {
    int e = blockIdx.x * blockDim.x + threadIdx.x;
    if (e < E) atomicAdd(&deg[row[e]], w[e]);
}

__global__ void k_dinv(float* deg, int n) {
    int i = blockIdx.x * blockDim.x + threadIdx.x;
    if (i < n) {
        float d = deg[i];
        deg[i] = d > 0.f ? rsqrtf(d) : 0.f;
    }
}

__global__ void k_norm(const int* __restrict__ row, const int* __restrict__ col,
                       const float* __restrict__ w, const float* __restrict__ dinv,
                       float* __restrict__ norm, int E) {
    int e = blockIdx.x * blockDim.x + threadIdx.x;
    if (e < E) norm[e] = -w[e] * dinv[row[e]] * dinv[col[e]];
}

// Acc[i] = bx[i%H] + bh[i%H]   (seed gate accumulators with both biases)
__global__ void k_bias(float* __restrict__ acc, const float* __restrict__ bx,
                       const float* __restrict__ bh, int H, int nh) {
    int i = blockIdx.x * blockDim.x + threadIdx.x;
    if (i < nh) {
        int c = i % H;
        acc[i] = bx[c] + bh[c];
    }
}

// ---------------- SpMM: Tout[col] += alpha * norm * Tin[row], float4 per thread
__global__ void k_spmm(const int* __restrict__ row, const int* __restrict__ col,
                       const float* __restrict__ norm,
                       const float* __restrict__ tin, float* __restrict__ tout,
                       int E, int F, float alpha) {
    int per = F >> 2;                  // float4 groups per edge
    int i = blockIdx.x * blockDim.x + threadIdx.x;
    if (i >= E * per) return;
    int e = i / per;
    int f = (i % per) << 2;
    float wv = alpha * norm[e];
    const float* src = tin + (size_t)row[e] * F + f;
    float4 v = *(const float4*)src;
    float* dst = tout + (size_t)col[e] * F + f;
    atomicAdd(dst + 0, wv * v.x);
    atomicAdd(dst + 1, wv * v.y);
    atomicAdd(dst + 2, wv * v.z);
    atomicAdd(dst + 3, wv * v.w);
}

// ---------------- W pre-pack into WMMA-B per-lane bf16 layout ----------------
// Packed block index: (((gsel*KCH + kcheb)*KSTEPS + ks)*NTILES + nt)
// Each block: 32 lanes x 16 bf16, so the GEMM kernel loads one contiguous v16bf
// per lane. B layout (bf16 32x16): lanes 0-15 K=kb+{0..7,16..23}, lanes 16-31 K=kb+{8..15,24..31},
// N = lane%16.
__global__ void k_pack_w(const float* __restrict__ W, __bf16* __restrict__ Bp,
                         int F, int H, int total) {
    int i = blockIdx.x * blockDim.x + threadIdx.x;
    if (i >= total) return;
    int ksteps = F >> 5, ntiles = H >> 4;
    int e2   = i & 15;
    int lane = (i >> 4) & 31;
    int blk  = i >> 9;
    int nt = blk % ntiles; blk /= ntiles;
    int ks = blk % ksteps; blk /= ksteps;
    int kcheb = blk % KCH;
    int gsel  = blk / KCH;
    int g = gsel * 2;                         // gates z (0) and h~ (2)
    int lmod = lane & 15, lhalf = lane >> 4;
    int wrow = (ks << 5) + 8 * lhalf + (e2 & 7) + ((e2 >> 3) << 4);
    int wcol = nt * 16 + lmod;
    Bp[i] = (__bf16)W[((size_t)(g * KCH + kcheb) * F + wrow) * H + wcol];
}

// ---------------- WMMA GEMM-accumulate -----------------
// Acc_g += A(N x F) @ W[g][kcheb] (F x H), gates g in {0 (z), 2 (h~)}.
// One wave owns one 16x16 output tile; block = 32 * 2 * (H/16) threads.
// A: f32 row-major, converted to bf16 in regs (v_cvt_pk_bf16_f32); accum f32.
// B: pre-packed bf16, one v16bf vector load per K-step.
template <int F, int H>
__global__ void k_cheb_gemm(const float* __restrict__ A, const __bf16* __restrict__ Bp,
                            float* __restrict__ Acc0, float* __restrict__ Acc2,
                            int kcheb, int Nn) {
    constexpr int KSTEPS = F / 32;
    constexpr int NTILES = H / 16;
    const int lane  = threadIdx.x & 31;
    const int wave  = threadIdx.x >> 5;
    const int gsel  = wave / NTILES;          // 0 -> gate 0 (z), 1 -> gate 2 (h~)
    const int nt    = wave % NTILES;
    const int lmod  = lane & 15;
    const int lhalf = lane >> 4;
    const int m16   = blockIdx.x << 4;
    float* Acc = gsel ? Acc2 : Acc0;
    const int colN = nt * 16 + lmod;
    const bool full = (m16 + 16 <= Nn);

    // C/D layout: VGPR j holds M = m16 + j + 8*lhalf, N = colN
    v8f c;
    if (full) {
#pragma unroll
        for (int j = 0; j < 8; ++j)
            c[j] = Acc[(size_t)(m16 + j + 8 * lhalf) * H + colN];
    } else {
#pragma unroll
        for (int j = 0; j < 8; ++j) {
            int r = m16 + j + 8 * lhalf;
            c[j] = (r < Nn) ? Acc[(size_t)r * H + colN] : 0.f;
        }
    }

    const int arow_i = m16 + lmod;
    const int arow   = arow_i < Nn ? arow_i : Nn - 1;
    // A bf16 16x32 layout: lane<16 K=kb+{0..7,16..23}; lane>=16 K=kb+{8..15,24..31}
    const float* abase = A + (size_t)arow * F + 8 * lhalf;
    const v16bf* bbase = (const v16bf*)Bp +
        ((size_t)(gsel * KCH + kcheb) * KSTEPS) * NTILES * 32 + nt * 32 + lane;

#pragma unroll
    for (int ks = 0; ks < KSTEPS; ++ks) {
        const float* ap = abase + (ks << 5);
        float4 a0 = *(const float4*)(ap + 0);
        float4 a1 = *(const float4*)(ap + 4);
        float4 a2 = *(const float4*)(ap + 16);
        float4 a3 = *(const float4*)(ap + 20);
        v16bf a;
        a[0]  = (__bf16)a0.x; a[1]  = (__bf16)a0.y; a[2]  = (__bf16)a0.z; a[3]  = (__bf16)a0.w;
        a[4]  = (__bf16)a1.x; a[5]  = (__bf16)a1.y; a[6]  = (__bf16)a1.z; a[7]  = (__bf16)a1.w;
        a[8]  = (__bf16)a2.x; a[9]  = (__bf16)a2.y; a[10] = (__bf16)a2.z; a[11] = (__bf16)a2.w;
        a[12] = (__bf16)a3.x; a[13] = (__bf16)a3.y; a[14] = (__bf16)a3.z; a[15] = (__bf16)a3.w;
        v16bf b = bbase[(size_t)ks * NTILES * 32];
        c = __builtin_amdgcn_wmma_f32_16x16x32_bf16(
                /*neg_a=*/false, a, /*neg_b=*/false, b,
                /*c_mod=*/(short)0, c, /*reuse_a=*/false, /*reuse_b=*/false);
    }

    if (full) {
#pragma unroll
        for (int j = 0; j < 8; ++j)
            Acc[(size_t)(m16 + j + 8 * lhalf) * H + colN] = c[j];
    } else {
#pragma unroll
        for (int j = 0; j < 8; ++j) {
            int r = m16 + j + 8 * lhalf;
            if (r < Nn) Acc[(size_t)r * H + colN] = c[j];
        }
    }
}

// h = relu((1 - sigmoid(Az)) * tanh(Ah))   (GRU with h_prev = 0, then ReLU)
__global__ void k_gate_relu(const float* __restrict__ Az, const float* __restrict__ Ah,
                            float* __restrict__ hout, int n) {
    int i = blockIdx.x * blockDim.x + threadIdx.x;
    if (i < n) {
        float z = 1.f / (1.f + expf(-Az[i]));
        float ht = tanhf(Ah[i]);
        hout[i] = fmaxf(0.f, (1.f - z) * ht);
    }
}

// logits = h2 @ Wlin + blin ; out = log_softmax(logits)
__global__ void k_head(const float* __restrict__ h, const float* __restrict__ Wl,
                       const float* __restrict__ bl, float* __restrict__ out, int N) {
    __shared__ float sw[HID2 * NCLS];
    __shared__ float sb[NCLS];
    int t = threadIdx.x;
    if (t < HID2 * NCLS) sw[t] = Wl[t];
    if (t < NCLS) sb[t] = bl[t];
    __syncthreads();
    int n = blockIdx.x * blockDim.x + t;
    if (n >= N) return;
    float hr[HID2];
#pragma unroll
    for (int k = 0; k < HID2; ++k) hr[k] = h[(size_t)n * HID2 + k];
    float l[NCLS];
    float m = -1e30f;
#pragma unroll
    for (int c = 0; c < NCLS; ++c) {
        float acc = sb[c];
#pragma unroll
        for (int k = 0; k < HID2; ++k) acc += hr[k] * sw[k * NCLS + c];
        l[c] = acc;
        m = fmaxf(m, acc);
    }
    float s = 0.f;
#pragma unroll
    for (int c = 0; c < NCLS; ++c) s += expf(l[c] - m);
    float lse = m + logf(s);
#pragma unroll
    for (int c = 0; c < NCLS; ++c) out[(size_t)n * NCLS + c] = l[c] - lse;
}

// ---------------- host side ----------------

struct Bufs {
    const int* row; const int* col;
    float* nrm;
    float* TA; float* TB; float* TC;
    float* Az; float* Ah;
    __bf16* Bp;
    int N, E;
};

static inline unsigned blks(long long n, int t) { return (unsigned)((n + t - 1) / t); }

// one GConvGRU layer with h_prev = 0 (only gates z and h~ matter)
template <int F, int H>
static void run_layer(const float* Xin, const float* Wx, const float* bx,
                      const float* bh, float* hout, const Bufs& B,
                      hipStream_t stream) {
    const int T = 256;
    const int N = B.N, E = B.E;
    const long long NH = (long long)N * H;
    const long long NF = (long long)N * F;
    const long long SP = (long long)E * (F >> 2);
    const unsigned gb = blks(N, 16);
    const int gt = 32 * 2 * (H / 16);
    const int pack_total = 2 * KCH * (F / 32) * (H / 16) * 32 * 16;

    k_pack_w<<<blks(pack_total, T), T, 0, stream>>>(Wx, B.Bp, F, H, pack_total);
    k_bias<<<blks(NH, T), T, 0, stream>>>(B.Az, bx + 0 * H, bh + 0 * H, H, (int)NH);
    k_bias<<<blks(NH, T), T, 0, stream>>>(B.Ah, bx + 2 * H, bh + 2 * H, H, (int)NH);

    // k = 0 : T0 = Xin (read in place)
    k_cheb_gemm<F, H><<<gb, gt, 0, stream>>>(Xin, B.Bp, B.Az, B.Ah, 0, N);
    // k = 1 : TA = L Xin
    k_prep<<<blks(NF, T), T, 0, stream>>>(B.TA, Xin, 0, (int)NF);
    k_spmm<<<blks(SP, T), T, 0, stream>>>(B.row, B.col, B.nrm, Xin, B.TA, E, F, 1.f);
    k_cheb_gemm<F, H><<<gb, gt, 0, stream>>>(B.TA, B.Bp, B.Az, B.Ah, 1, N);
    // k = 2 : TB = 2 L TA - Xin
    k_prep<<<blks(NF, T), T, 0, stream>>>(B.TB, Xin, 1, (int)NF);
    k_spmm<<<blks(SP, T), T, 0, stream>>>(B.row, B.col, B.nrm, B.TA, B.TB, E, F, 2.f);
    k_cheb_gemm<F, H><<<gb, gt, 0, stream>>>(B.TB, B.Bp, B.Az, B.Ah, 2, N);
    // k = 3 : TC = 2 L TB - TA
    k_prep<<<blks(NF, T), T, 0, stream>>>(B.TC, B.TA, 1, (int)NF);
    k_spmm<<<blks(SP, T), T, 0, stream>>>(B.row, B.col, B.nrm, B.TB, B.TC, E, F, 2.f);
    k_cheb_gemm<F, H><<<gb, gt, 0, stream>>>(B.TC, B.Bp, B.Az, B.Ah, 3, N);
    // k = 4 : TA = 2 L TC - TB   (TA no longer needed, reuse)
    k_prep<<<blks(NF, T), T, 0, stream>>>(B.TA, B.TB, 1, (int)NF);
    k_spmm<<<blks(SP, T), T, 0, stream>>>(B.row, B.col, B.nrm, B.TC, B.TA, E, F, 2.f);
    k_cheb_gemm<F, H><<<gb, gt, 0, stream>>>(B.TA, B.Bp, B.Az, B.Ah, 4, N);

    k_gate_relu<<<blks(NH, T), T, 0, stream>>>(B.Az, B.Ah, hout, (int)NH);
}

extern "C" void kernel_launch(void* const* d_in, const int* in_sizes, int n_in,
                              void* d_out, int out_size, void* d_ws, size_t ws_size,
                              hipStream_t stream) {
    const float* x    = (const float*)d_in[0];
    const float* ew   = (const float*)d_in[1];
    const float* Wx1  = (const float*)d_in[2];
    const float* bx1  = (const float*)d_in[4];
    const float* bh1  = (const float*)d_in[5];
    const float* Wx2  = (const float*)d_in[6];
    const float* bx2  = (const float*)d_in[8];
    const float* bh2  = (const float*)d_in[9];
    const float* Wlin = (const float*)d_in[10];
    const float* blin = (const float*)d_in[11];
    const int*   ei   = (const int*)d_in[12];

    const int N = in_sizes[0] / F_IN;
    const int E = in_sizes[1];

    // workspace carve-up (floats; keep 32B alignment for the packed-B region)
    float* ws = (float*)d_ws;
    size_t off = 0;
    float* dinv = ws + off; off += (size_t)N;        off = (off + 7) & ~(size_t)7;
    float* nrm  = ws + off; off += (size_t)E;
    float* TA   = ws + off; off += (size_t)N * F_IN;
    float* TB   = ws + off; off += (size_t)N * F_IN;
    float* TC   = ws + off; off += (size_t)N * F_IN;
    float* Az   = ws + off; off += (size_t)N * HID1;
    float* Ah   = ws + off; off += (size_t)N * HID1;
    float* h1   = ws + off; off += (size_t)N * HID1;
    float* h2   = ws + off; off += (size_t)N * HID2; off = (off + 7) & ~(size_t)7;
    __bf16* Bp  = (__bf16*)(ws + off);               // <= 80KB packed weights

    const int T = 256;

    // --- Chebyshev normalization: L_hat = -D^-1/2 A D^-1/2 ---
    const int* row = ei;
    const int* col = ei + E;
    k_zero<<<blks(N, T), T, 0, stream>>>(dinv, N);
    k_deg <<<blks(E, T), T, 0, stream>>>(row, ew, dinv, E);
    k_dinv<<<blks(N, T), T, 0, stream>>>(dinv, N);
    k_norm<<<blks(E, T), T, 0, stream>>>(row, col, ew, dinv, nrm, E);

    Bufs B{row, col, nrm, TA, TB, TC, Az, Ah, Bp, N, E};
    run_layer<F_IN, HID1>(x,  Wx1, bx1, bh1, h1, B, stream);
    run_layer<HID1, HID2>(h1, Wx2, bx2, bh2, h2, B, stream);

    k_head<<<blks(N, T), T, 0, stream>>>(h2, Wlin, blin, (float*)d_out, N);
}
